// VectorQuantizer_22814866276990
// MI455X (gfx1250) — matrix operand order repaired and checked by reading, not compiled, not run on hardware
//
#include <hip/hip_runtime.h>
#include <math.h>

typedef __attribute__((ext_vector_type(16))) _Float16 v16h;
typedef __attribute__((ext_vector_type(8)))  float    v8f;

#define NUM_EMB 512
#define EMB_DIM 64
#define NPOS    8192
#define N_VEC   (32 * 8192)
#define TOTAL_ELEMS (N_VEC * EMB_DIM)

// workspace layout (bytes)
#define HIST_BYTES  ((size_t)NPOS * NUM_EMB * sizeof(int))          // 16 MB
#define WHALF_OFF   (HIST_BYTES)                                     // 64 KB of f16 codebook
#define WNORM_OFF   (WHALF_OFF + (size_t)NUM_EMB * EMB_DIM * sizeof(_Float16))
#define LOSSP_OFF   (WNORM_OFF + (size_t)NUM_EMB * sizeof(float))
#define ENTP_OFF    (LOSSP_OFF + 2048 * sizeof(float))

// ---------------------------------------------------------------- zero hist
__global__ __launch_bounds__(256) void vq_zero(int* __restrict__ hist) {
    const size_t i = ((size_t)blockIdx.x * 256 + threadIdx.x) * 2;   // int4 pairs
    int4 z = {0, 0, 0, 0};
    ((int4*)hist)[i]     = z;
    ((int4*)hist)[i + 1] = z;
}

// --------------------------------------------- codebook -> f16 + squared norms
__global__ __launch_bounds__(256) void vq_prep(const float* __restrict__ W,
                                               _Float16* __restrict__ Wh,
                                               float* __restrict__ Wn) {
    const int c = blockIdx.x * 256 + threadIdx.x;
    if (c >= NUM_EMB) return;
    float n = 0.f;
    #pragma unroll 8
    for (int k = 0; k < EMB_DIM; ++k) {
        float w = W[c * EMB_DIM + k];
        n += w * w;
        Wh[c * EMB_DIM + k] = (_Float16)w;
    }
    Wn[c] = n;
}

// ------------------------------------------- main: WMMA distances + argmin +
//                                             gather + loss partial + histogram
__global__ __launch_bounds__(256)
void vq_main(const float* __restrict__ X, const float* __restrict__ W,
             const _Float16* __restrict__ Wh, const float* __restrict__ Wn,
             float* __restrict__ OUT, int* __restrict__ hist,
             float* __restrict__ lossPart) {
    __shared__ int   sIdx[8][16];
    __shared__ float sRed[256];

    const int tid  = threadIdx.x;
    const int wave = tid >> 5;
    const int lane = tid & 31;
    const int m    = lane & 15;      // row (for A) / column-in-tile (for B,C)
    const int hsel = lane >> 4;      // which K/M sub-half this lane holds
    const int rowBase = (blockIdx.x * 8 + wave) * 16;

    // ---- A fragments: 16x32 f16, contiguous layout (lane m holds K = hsel*16 + {0..15})
    const float4* x4 = (const float4*)(X + (size_t)(rowBase + m) * EMB_DIM);
    float4 s0 = x4[hsel * 4 + 0],  s1 = x4[hsel * 4 + 1];       // K hsel*16 + 0..7
    float4 s2 = x4[hsel * 4 + 2],  s3 = x4[hsel * 4 + 3];       // K hsel*16 + 8..15
    float4 s4 = x4[8 + hsel * 4],  s5 = x4[9 + hsel * 4];       // K 32 + hsel*16 + 0..7
    float4 s6 = x4[10 + hsel * 4], s7 = x4[11 + hsel * 4];      // K 32 + hsel*16 + 8..15
    v16h a0, a1;
    a0[0]=(_Float16)s0.x; a0[1]=(_Float16)s0.y; a0[2]=(_Float16)s0.z; a0[3]=(_Float16)s0.w;
    a0[4]=(_Float16)s1.x; a0[5]=(_Float16)s1.y; a0[6]=(_Float16)s1.z; a0[7]=(_Float16)s1.w;
    a0[8]=(_Float16)s2.x; a0[9]=(_Float16)s2.y; a0[10]=(_Float16)s2.z; a0[11]=(_Float16)s2.w;
    a0[12]=(_Float16)s3.x; a0[13]=(_Float16)s3.y; a0[14]=(_Float16)s3.z; a0[15]=(_Float16)s3.w;
    a1[0]=(_Float16)s4.x; a1[1]=(_Float16)s4.y; a1[2]=(_Float16)s4.z; a1[3]=(_Float16)s4.w;
    a1[4]=(_Float16)s5.x; a1[5]=(_Float16)s5.y; a1[6]=(_Float16)s5.z; a1[7]=(_Float16)s5.w;
    a1[8]=(_Float16)s6.x; a1[9]=(_Float16)s6.y; a1[10]=(_Float16)s6.z; a1[11]=(_Float16)s6.w;
    a1[12]=(_Float16)s7.x; a1[13]=(_Float16)s7.y; a1[14]=(_Float16)s7.z; a1[15]=(_Float16)s7.w;

    float minv[8];
    int   mini[8];
    #pragma unroll
    for (int r = 0; r < 8; ++r) { minv[r] = 3.4e38f; mini[r] = 0; }

    // ---- 32 column tiles of 16 codes; 2 chained WMMAs per tile (K=64)
    #pragma unroll 4
    for (int t = 0; t < 32; ++t) {
        const int code = t * 16 + m;
        const _Float16* wr = Wh + (size_t)code * EMB_DIM + hsel * 16;
        v16h b0 = *(const v16h*)(wr);         // K = 0..31 half per B layout
        v16h b1 = *(const v16h*)(wr + 32);    // K = 32..63 half
        v8f acc = {};
        acc = __builtin_amdgcn_wmma_f32_16x16x32_f16(false, a0, false, b0,
                                                     (short)0, acc, false, false);
        acc = __builtin_amdgcn_wmma_f32_16x16x32_f16(false, a1, false, b1,
                                                     (short)0, acc, false, false);
        const float wn = Wn[code];
        #pragma unroll
        for (int r = 0; r < 8; ++r) {
            float s = wn - 2.0f * acc[r];     // dist minus constant ||x||^2
            if (s < minv[r]) { minv[r] = s; mini[r] = code; }
        }
    }

    // ---- argmin across the 16 lanes sharing a row (xor stays inside each half)
    #pragma unroll
    for (int r = 0; r < 8; ++r) {
        float v = minv[r];
        int   i = mini[r];
        #pragma unroll
        for (int off = 8; off >= 1; off >>= 1) {
            float vo = __shfl_xor(v, off, 32);
            int   io = __shfl_xor(i, off, 32);
            if (vo < v || (vo == v && io < i)) { v = vo; i = io; }  // first-min tiebreak
        }
        if (m == 0) sIdx[wave][r + 8 * hsel] = i;   // row = r + 8*hsel
    }
    __syncthreads();

    // ---- gather f32 codes, write output, loss partial, histogram
    float lsum = 0.f;
    for (int rr = 0; rr < 16; ++rr) {
        const int row  = rowBase + rr;
        const int code = sIdx[wave][rr];
        const float2 wv = ((const float2*)(W + (size_t)code * EMB_DIM))[lane];
        const float2 xv = ((const float2*)(X + (size_t)row * EMB_DIM))[lane];
        ((float2*)(OUT + (size_t)row * EMB_DIM))[lane] = wv;   // ST forward == quantized
        float d0 = wv.x - xv.x, d1 = wv.y - xv.y;
        lsum += d0 * d0 + d1 * d1;
        if (lane == 0)
            atomicAdd(hist + (size_t)(row & (NPOS - 1)) * NUM_EMB + code, 1);
    }

    sRed[tid] = lsum;
    __syncthreads();
    for (int s = 128; s > 0; s >>= 1) {
        if (tid < s) sRed[tid] += sRed[tid + s];
        __syncthreads();
    }
    if (tid == 0) lossPart[blockIdx.x] = sRed[0];
}

// ------------------------------------------------ entropy partials over histogram
__global__ __launch_bounds__(256)
void vq_entropy(const int* __restrict__ hist, float* __restrict__ entPart) {
    __shared__ float sRed[256];
    const int tid  = threadIdx.x;
    const size_t base = (size_t)blockIdx.x * 4096;
    float s = 0.f;
    #pragma unroll 4
    for (int j = 0; j < 16; ++j) {
        int c = hist[base + j * 256 + tid];
        if (c > 0) {
            float p = (float)c * (1.0f / 32.0f);
            s += p * logf(p + 1e-10f);
        }
    }
    sRed[tid] = s;
    __syncthreads();
    for (int k = 128; k > 0; k >>= 1) {
        if (tid < k) sRed[tid] += sRed[tid + k];
        __syncthreads();
    }
    if (tid == 0) entPart[blockIdx.x] = sRed[0];
}

// ------------------------------------------------ final scalars
__global__ __launch_bounds__(256)
void vq_final(const float* __restrict__ lossPart, const float* __restrict__ entPart,
              float* __restrict__ out) {
    __shared__ float sL[256], sE[256];
    const int tid = threadIdx.x;
    float l = 0.f, e = 0.f;
    for (int i = tid; i < 2048; i += 256) l += lossPart[i];
    for (int i = tid; i < 1024; i += 256) e += entPart[i];
    sL[tid] = l; sE[tid] = e;
    __syncthreads();
    for (int s = 128; s > 0; s >>= 1) {
        if (tid < s) { sL[tid] += sL[tid + s]; sE[tid] += sE[tid + s]; }
        __syncthreads();
    }
    if (tid == 0) {
        out[0] = 1.25f * sL[0] / (float)TOTAL_ELEMS;  // q_loss + 0.25*e_loss
        out[1] = expf(-sE[0]);                        // perplexity
    }
}

extern "C" void kernel_launch(void* const* d_in, const int* in_sizes, int n_in,
                              void* d_out, int out_size, void* d_ws, size_t ws_size,
                              hipStream_t stream) {
    const float* X = (const float*)d_in[0];   // [32, 8192, 64] f32
    const float* W = (const float*)d_in[1];   // [512, 64] f32
    float* OUT = (float*)d_out;               // 16777216 quantized + loss + perplexity

    char* ws = (char*)d_ws;
    int*      hist  = (int*)(ws);
    _Float16* Wh    = (_Float16*)(ws + WHALF_OFF);
    float*    Wn    = (float*)(ws + WNORM_OFF);
    float*    lossP = (float*)(ws + LOSSP_OFF);
    float*    entP  = (float*)(ws + ENTP_OFF);

    vq_zero   <<<2048, 256, 0, stream>>>(hist);
    vq_prep   <<<2,    256, 0, stream>>>(W, Wh, Wn);
    vq_main   <<<2048, 256, 0, stream>>>(X, W, Wh, Wn, OUT, hist, lossP);
    vq_entropy<<<1024, 256, 0, stream>>>(hist, entP);
    vq_final  <<<1,    256, 0, stream>>>(lossP, entP, OUT + TOTAL_ELEMS);
}
